// MoeMegaBlocks_52982716563635
// MI455X (gfx1250) — compile-verified, matrix operand order checked by reading, not run on hardware
//
#include <hip/hip_runtime.h>
#include <hip/hip_bf16.h>
#include <math.h>

// ---------------------------------------------------------------------------
// MoE (top-2 of 8 experts), D=1024, F=2048, N=8192 tokens.
// bf16 WMMA grouped GEMMs + gfx1250 async global->LDS pipeline (ASYNCcnt).
// Workspace (~151.3 MB):
//   hb  : 2N*F bf16          (67.1 MB)
//   xb  : N*D bf16           (16.8 MB)
//   w1b : D*E*F bf16         (33.6 MB)
//   w2b : E*F*D bf16         (33.6 MB)
//   row_ids/row_w/topk_id/topk_w : 4 * 2N*4
//   cnt[8], off[8], cursor[8]
// ---------------------------------------------------------------------------

typedef __bf16 bf16_t;
typedef __attribute__((ext_vector_type(16))) __bf16 v16bf;
typedef __attribute__((ext_vector_type(8)))  __bf16 v8bf;
typedef __attribute__((ext_vector_type(4)))  __bf16 v4bf;
typedef __attribute__((ext_vector_type(8)))  float  v8f;

#define D_MODEL 1024
#define D_FFN   2048
#define N_EXP   8
#define LDSTR   40   // 32 K-elems + 8 pad (16B-aligned rows, skewed banks)

__device__ __forceinline__ bf16_t f2bf(float f) { return (bf16_t)f; }

// Async 16B global->LDS copy (per lane). LDS byte address = low 32 bits of the
// flat pointer (ISA 10.2: flat-LDS aperture keeps offset in addr[31:0]).
__device__ __forceinline__ void async_b128(const void* g, const void* l) {
  unsigned loff = (unsigned)(unsigned long long)l;
  asm volatile("global_load_async_to_lds_b128 %0, %1, off"
               :: "v"(loff), "v"(g) : "memory");
}
#define WAIT_ASYNC(n) asm volatile("s_wait_asynccnt " #n ::: "memory")

// v16bf WMMA operand fragment from an LDS row (LDSTR bf16 stride).
// lanes 0-15: K = kb..kb+7 , 16+kb..16+kb+7 with kb = 8*(lane>>4).
__device__ __forceinline__ v16bf load_frag(const bf16_t* row, int lane) {
  int kb = (lane >> 4) << 3;
  v8bf lo = *(const v8bf*)(row + kb);
  v8bf hi = *(const v8bf*)(row + 16 + kb);
  v16bf f;
#pragma unroll
  for (int j = 0; j < 8; ++j) { f[j] = lo[j]; f[j + 8] = hi[j]; }
  return f;
}

// ------------------------- fp32 -> bf16 pre-convert ------------------------
__global__ __launch_bounds__(256) void cvt_bf16(
    const float* __restrict__ in, bf16_t* __restrict__ out, long n4)
{
  long i = (long)blockIdx.x * 256 + threadIdx.x;
  if (i >= n4) return;
  const float4 v = ((const float4*)in)[i];
  v4bf o;
  o[0] = f2bf(v.x); o[1] = f2bf(v.y); o[2] = f2bf(v.z); o[3] = f2bf(v.w);
  ((v4bf*)out)[i] = o;
}

// ------------------------------ router -------------------------------------
__global__ __launch_bounds__(256) void moe_router(
    const float* __restrict__ x, const float* __restrict__ rw,
    int* __restrict__ topk_id, float* __restrict__ topk_w,
    int* __restrict__ cnt, int ntok)
{
  __shared__ float rws[N_EXP * D_MODEL];
  for (int i = threadIdx.x; i < N_EXP * D_MODEL; i += 256) rws[i] = rw[i];
  __syncthreads();

  int wave = threadIdx.x >> 5, lane = threadIdx.x & 31;
  int tok = blockIdx.x * 8 + wave;
  if (tok >= ntok) return;

  float acc[N_EXP];
#pragma unroll
  for (int e = 0; e < N_EXP; ++e) acc[e] = 0.f;
  const float* xr = x + (size_t)tok * D_MODEL;
  for (int d = lane; d < D_MODEL; d += 32) {
    float xv = xr[d];
#pragma unroll
    for (int e = 0; e < N_EXP; ++e) acc[e] += xv * rws[e * D_MODEL + d];
  }
#pragma unroll
  for (int e = 0; e < N_EXP; ++e) {
#pragma unroll
    for (int off = 16; off > 0; off >>= 1)
      acc[e] += __shfl_xor(acc[e], off, 32);
  }
  if (lane == 0) {
    float mx = acc[0];
#pragma unroll
    for (int e = 1; e < N_EXP; ++e) mx = fmaxf(mx, acc[e]);
    float p[N_EXP]; float s = 0.f;
#pragma unroll
    for (int e = 0; e < N_EXP; ++e) { p[e] = __expf(acc[e] - mx); s += p[e]; }
    float inv_s = 1.f / s;
#pragma unroll
    for (int e = 0; e < N_EXP; ++e) p[e] *= inv_s;
    int i0 = 0; float v0 = p[0];
#pragma unroll
    for (int e = 1; e < N_EXP; ++e) if (p[e] > v0) { v0 = p[e]; i0 = e; }
    int i1 = (i0 == 0) ? 1 : 0; float v1 = p[i1];
#pragma unroll
    for (int e = 0; e < N_EXP; ++e)
      if (e != i0 && p[e] > v1) { v1 = p[e]; i1 = e; }
    float inv = 1.f / (v0 + v1);
    topk_id[tok * 2 + 0] = i0; topk_w[tok * 2 + 0] = v0 * inv;
    topk_id[tok * 2 + 1] = i1; topk_w[tok * 2 + 1] = v1 * inv;
    atomicAdd(&cnt[i0], 1);
    atomicAdd(&cnt[i1], 1);
  }
}

__global__ void moe_scan(const int* __restrict__ cnt, int* __restrict__ off) {
  if (threadIdx.x == 0) {
    int a = 0;
    for (int e = 0; e < N_EXP; ++e) { off[e] = a; a += cnt[e]; }
  }
}

__global__ __launch_bounds__(256) void moe_scatter(
    const int* __restrict__ topk_id, const float* __restrict__ topk_w,
    const int* __restrict__ off, int* __restrict__ cursor,
    int* __restrict__ row_ids, float* __restrict__ row_w, int ntok)
{
  int n = blockIdx.x * 256 + threadIdx.x;
  if (n >= ntok) return;
#pragma unroll
  for (int s = 0; s < 2; ++s) {
    int e = topk_id[n * 2 + s];
    int p = atomicAdd(&cursor[e], 1);
    int pos = off[e] + p;
    row_ids[pos] = n;
    row_w[pos]   = topk_w[n * 2 + s];
  }
}

// ------------------------- GEMM1: H = gelu(Xb @ W1b_e) ---------------------
// 256 thr = 8 waves (4x2); per wave 32x64 (2x4 WMMA accums). K-step 32.
// A tiles: async global->LDS (double buffered); B tiles: b128 reg loads +
// transposed ds stores (double buffered).
__global__ __launch_bounds__(256) void moe_ffn1(
    const bf16_t* __restrict__ xb, const bf16_t* __restrict__ w1b,
    const int* __restrict__ row_ids, const int* __restrict__ cnt,
    const int* __restrict__ off, bf16_t* __restrict__ h)
{
  int e = blockIdx.y;
  int cntE = cnt[e];
  int mBase = blockIdx.x * 128;
  if (mBase >= cntE) return;
  int offE = off[e];
  int fBase = blockIdx.z * 128;

  __shared__ alignas(16) bf16_t As[2][128 * LDSTR];
  __shared__ alignas(16) bf16_t Bs[2][128 * LDSTR];   // [col][k]
  __shared__ int ids[128];

  int t = threadIdx.x;
  if (t < 128) {
    int r = mBase + t;
    ids[t] = (r < cntE) ? row_ids[offE + r] : 0;
  }
  __syncthreads();

  int wave = t >> 5, lane = t & 31;
  int wm = wave & 3, wn = wave >> 2;
  int sub = lane & 15;

  // Per-thread A-async assignment: 2 chunks of 16B; chunk j covers row j*64+(t>>2).
  int rA = t >> 2, kcA = (t & 3) * 8;                   // kcA in elements
  const bf16_t* gA0 = xb + (size_t)ids[rA]      * D_MODEL + kcA;
  const bf16_t* gA1 = xb + (size_t)ids[64 + rA] * D_MODEL + kcA;
  bf16_t* lA0[2]; bf16_t* lA1[2];
#pragma unroll
  for (int b = 0; b < 2; ++b) {
    lA0[b] = &As[b][rA * LDSTR + kcA];
    lA1[b] = &As[b][(64 + rA) * LDSTR + kcA];
  }
  const bf16_t* w1be = w1b + (size_t)e * D_FFN + fBase;

  v8f c[2][4];
#pragma unroll
  for (int mi = 0; mi < 2; ++mi)
#pragma unroll
    for (int ni = 0; ni < 4; ++ni) { v8f z = {}; c[mi][ni] = z; }

  // prologue: tile 0 A-async into buffer 0 (2 issues per wave)
  async_b128(gA0, lA0[0]);
  async_b128(gA1, lA1[0]);

  for (int k0 = 0, it = 0; k0 < D_MODEL; k0 += 32, ++it) {
    int cur = it & 1;
    // B(k0) -> regs (2 x b128)
    v8bf br[2]; int kk[2], cc[2];
#pragma unroll
    for (int j = 0; j < 2; ++j) {
      int ch = j * 256 + t;
      kk[j] = ch >> 4; cc[j] = (ch & 15) * 8;
      br[j] = *(const v8bf*)(w1be + (size_t)(k0 + kk[j]) * (D_FFN * N_EXP) + cc[j]);
    }
    // prefetch next A tile asynchronously into the other buffer
    if (k0 + 32 < D_MODEL) {
      async_b128(gA0 + k0 + 32, lA0[cur ^ 1]);
      async_b128(gA1 + k0 + 32, lA1[cur ^ 1]);
    }
    // transpose-store B into LDS [col][k]
#pragma unroll
    for (int j = 0; j < 2; ++j)
#pragma unroll
      for (int q = 0; q < 8; ++q)
        Bs[cur][(cc[j] + q) * LDSTR + kk[j]] = br[j][q];
    if (k0 + 32 < D_MODEL) { WAIT_ASYNC(2); } else { WAIT_ASYNC(0); }
    __syncthreads();

    v16bf a[2], b[4];
#pragma unroll
    for (int mi = 0; mi < 2; ++mi)
      a[mi] = load_frag(&As[cur][(wm * 32 + mi * 16 + sub) * LDSTR], lane);
#pragma unroll
    for (int ni = 0; ni < 4; ++ni)
      b[ni] = load_frag(&Bs[cur][(wn * 64 + ni * 16 + sub) * LDSTR], lane);
#pragma unroll
    for (int mi = 0; mi < 2; ++mi)
#pragma unroll
      for (int ni = 0; ni < 4; ++ni)
        c[mi][ni] = __builtin_amdgcn_wmma_f32_16x16x32_bf16(
            false, a[mi], false, b[ni], (short)0, c[mi][ni], false, false);
    __syncthreads();
  }

  int rsub = (lane >> 4) << 3;   // C layout: VGPR j -> M = j + 8*(lane>>4)
#pragma unroll
  for (int mi = 0; mi < 2; ++mi) {
#pragma unroll
    for (int ni = 0; ni < 4; ++ni) {
      int colg = fBase + wn * 64 + ni * 16 + sub;
#pragma unroll
      for (int j = 0; j < 8; ++j) {
        int r = wm * 32 + mi * 16 + rsub + j;
        if (mBase + r < cntE) {
          float v = c[mi][ni][j];
          float g = 0.5f * v * (1.0f + erff(v * 0.70710678118654752f));
          h[(size_t)(offE + mBase + r) * D_FFN + colg] = f2bf(g);
        }
      }
    }
  }
}

// --------------- GEMM2: out[tok] += w * (H_e @ W2b_e) ----------------------
__global__ __launch_bounds__(256) void moe_ffn2(
    const bf16_t* __restrict__ h, const bf16_t* __restrict__ w2b,
    const int* __restrict__ row_ids, const float* __restrict__ row_w,
    const int* __restrict__ cnt, const int* __restrict__ off,
    float* __restrict__ out, int ntok)
{
  int e = blockIdx.y;
  int cntE = cnt[e];
  int mBase = blockIdx.x * 128;
  if (mBase >= cntE) return;
  int offE = off[e];
  int nBase = blockIdx.z * 128;
  int maxSlot = 2 * ntok - 1;

  __shared__ alignas(16) bf16_t As[2][128 * LDSTR];
  __shared__ alignas(16) bf16_t Bs[2][128 * LDSTR];
  __shared__ int   ids[128];
  __shared__ float wgt[128];

  int t = threadIdx.x;
  if (t < 128) {
    int r = mBase + t;
    bool ok = r < cntE;
    ids[t] = ok ? row_ids[offE + r] : 0;
    wgt[t] = ok ? row_w[offE + r] : 0.f;
  }
  __syncthreads();

  int wave = t >> 5, lane = t & 31;
  int wm = wave & 3, wn = wave >> 2;
  int sub = lane & 15;

  // A rows are slot-indexed in H directly (no gather needed for loads).
  int rA = t >> 2, kcA = (t & 3) * 8;
  int slot0 = offE + mBase + rA;      if (slot0 > maxSlot) slot0 = maxSlot;
  int slot1 = offE + mBase + 64 + rA; if (slot1 > maxSlot) slot1 = maxSlot;
  const bf16_t* gA0 = h + (size_t)slot0 * D_FFN + kcA;
  const bf16_t* gA1 = h + (size_t)slot1 * D_FFN + kcA;
  bf16_t* lA0[2]; bf16_t* lA1[2];
#pragma unroll
  for (int b = 0; b < 2; ++b) {
    lA0[b] = &As[b][rA * LDSTR + kcA];
    lA1[b] = &As[b][(64 + rA) * LDSTR + kcA];
  }
  const bf16_t* w2be = w2b + (size_t)e * D_FFN * D_MODEL + nBase;

  v8f c[2][4];
#pragma unroll
  for (int mi = 0; mi < 2; ++mi)
#pragma unroll
    for (int ni = 0; ni < 4; ++ni) { v8f z = {}; c[mi][ni] = z; }

  async_b128(gA0, lA0[0]);
  async_b128(gA1, lA1[0]);

  for (int k0 = 0, it = 0; k0 < D_FFN; k0 += 32, ++it) {
    int cur = it & 1;
    v8bf br[2]; int kk[2], cc[2];
#pragma unroll
    for (int j = 0; j < 2; ++j) {
      int ch = j * 256 + t;
      kk[j] = ch >> 4; cc[j] = (ch & 15) * 8;
      br[j] = *(const v8bf*)(w2be + (size_t)(k0 + kk[j]) * D_MODEL + cc[j]);
    }
    if (k0 + 32 < D_FFN) {
      async_b128(gA0 + k0 + 32, lA0[cur ^ 1]);
      async_b128(gA1 + k0 + 32, lA1[cur ^ 1]);
    }
#pragma unroll
    for (int j = 0; j < 2; ++j)
#pragma unroll
      for (int q = 0; q < 8; ++q)
        Bs[cur][(cc[j] + q) * LDSTR + kk[j]] = br[j][q];
    if (k0 + 32 < D_FFN) { WAIT_ASYNC(2); } else { WAIT_ASYNC(0); }
    __syncthreads();

    v16bf a[2], b[4];
#pragma unroll
    for (int mi = 0; mi < 2; ++mi)
      a[mi] = load_frag(&As[cur][(wm * 32 + mi * 16 + sub) * LDSTR], lane);
#pragma unroll
    for (int ni = 0; ni < 4; ++ni)
      b[ni] = load_frag(&Bs[cur][(wn * 64 + ni * 16 + sub) * LDSTR], lane);
#pragma unroll
    for (int mi = 0; mi < 2; ++mi)
#pragma unroll
      for (int ni = 0; ni < 4; ++ni)
        c[mi][ni] = __builtin_amdgcn_wmma_f32_16x16x32_bf16(
            false, a[mi], false, b[ni], (short)0, c[mi][ni], false, false);
    __syncthreads();
  }

  int rsub = (lane >> 4) << 3;
#pragma unroll
  for (int mi = 0; mi < 2; ++mi) {
#pragma unroll
    for (int ni = 0; ni < 4; ++ni) {
      int colg = nBase + wn * 64 + ni * 16 + sub;
#pragma unroll
      for (int j = 0; j < 8; ++j) {
        int r = wm * 32 + mi * 16 + rsub + j;
        if (mBase + r < cntE) {
          float v = c[mi][ni][j] * wgt[r];
          atomicAdd(&out[(size_t)ids[r] * D_MODEL + colg], v);
        }
      }
    }
  }
}

// ------------------------------- host --------------------------------------
extern "C" void kernel_launch(void* const* d_in, const int* in_sizes, int n_in,
                              void* d_out, int out_size, void* d_ws, size_t ws_size,
                              hipStream_t stream) {
  const float* x  = (const float*)d_in[0];
  const float* rw = (const float*)d_in[1];
  const float* w1 = (const float*)d_in[2];
  const float* w2 = (const float*)d_in[3];
  float* out = (float*)d_out;

  int ntok  = in_sizes[0] / D_MODEL;   // 8192
  int nslot = 2 * ntok;
  long nx  = (long)ntok * D_MODEL;
  long nw1 = (long)D_MODEL * D_FFN * N_EXP;
  long nw2 = (long)D_FFN * N_EXP * D_MODEL;

  char* ws = (char*)d_ws;
  bf16_t* hb  = (bf16_t*)ws;
  size_t  o   = (size_t)nslot * D_FFN * sizeof(bf16_t);
  bf16_t* xb  = (bf16_t*)(ws + o);  o += (size_t)nx  * sizeof(bf16_t);
  bf16_t* w1b = (bf16_t*)(ws + o);  o += (size_t)nw1 * sizeof(bf16_t);
  bf16_t* w2b = (bf16_t*)(ws + o);  o += (size_t)nw2 * sizeof(bf16_t);
  int*   row_ids = (int*)  (ws + o); o += (size_t)nslot * 4;
  float* row_w   = (float*)(ws + o); o += (size_t)nslot * 4;
  int*   topk_id = (int*)  (ws + o); o += (size_t)nslot * 4;
  float* topk_w  = (float*)(ws + o); o += (size_t)nslot * 4;
  int*   cnt     = (int*)  (ws + o);
  int*   off     = cnt + 8;
  int*   cursor  = cnt + 16;

  hipMemsetAsync(cnt, 0, 24 * sizeof(int), stream);
  hipMemsetAsync(d_out, 0, (size_t)out_size * sizeof(float), stream);

  cvt_bf16<<<(unsigned)((nx  / 4 + 255) / 256), 256, 0, stream>>>(x,  xb,  nx  / 4);
  cvt_bf16<<<(unsigned)((nw1 / 4 + 255) / 256), 256, 0, stream>>>(w1, w1b, nw1 / 4);
  cvt_bf16<<<(unsigned)((nw2 / 4 + 255) / 256), 256, 0, stream>>>(w2, w2b, nw2 / 4);

  moe_router<<<(ntok + 7) / 8, 256, 0, stream>>>(x, rw, topk_id, topk_w, cnt, ntok);
  moe_scan<<<1, 32, 0, stream>>>(cnt, off);
  moe_scatter<<<(ntok + 255) / 256, 256, 0, stream>>>(topk_id, topk_w, off,
                                                      cursor, row_ids, row_w, ntok);

  int mTiles = (ntok + 127) / 128;
  dim3 g1(mTiles, N_EXP, D_FFN / 128);
  moe_ffn1<<<g1, 256, 0, stream>>>(xb, w1b, row_ids, cnt, off, hb);
  dim3 g2(mTiles, N_EXP, D_MODEL / 128);
  moe_ffn2<<<g2, 256, 0, stream>>>(hb, w2b, row_ids, row_w, cnt, off, out, ntok);
}
